// TripletLoss_53360673686075
// MI455X (gfx1250) — compile-verified, matrix operand order checked by reading, not run on hardware
//
#include <hip/hip_runtime.h>

typedef __attribute__((ext_vector_type(16))) __bf16 v16bf;
typedef __attribute__((ext_vector_type(8)))  float  v8f;

#define BB     8192
#define DD     256
#define NTILE  (BB / 16)
#define NSTRIP (NTILE / 2)      // blocks in pass kernels: strips {b, 511-b}
#define MARGINF 0.3f
#define BIGV    1e30f

union FragU { v16bf v; unsigned u[8]; };

static __device__ __forceinline__ v8f wmma_bf16(v16bf a, v16bf b, v8f c) {
  // 8 args: (neg_a, A, neg_b, B, c_mod, C, reuse_a, reuse_b)
  return __builtin_amdgcn_wmma_f32_16x16x32_bf16(false, a, false, b, (short)0, c,
                                                 false, false);
}

// fp32 -> bf16 bits, round-to-nearest-even (pure integer, no __bf16 scalar math)
static __device__ __forceinline__ unsigned short f32_to_bf16_rne(float x) {
  unsigned b = __float_as_uint(x);
  unsigned r = b + 0x7FFFu + ((b >> 16) & 1u);
  return (unsigned short)(r >> 16);
}

// order-preserving float <-> uint bijection (atomic max/min on uint == float order)
static __device__ __forceinline__ unsigned f2ord(float x) {
  unsigned b = __float_as_uint(x);
  return (b & 0x80000000u) ? ~b : (b | 0x80000000u);
}
static __device__ __forceinline__ float ord2f(unsigned u) {
  unsigned b = (u & 0x80000000u) ? (u & 0x7FFFFFFFu) : ~u;
  return __uint_as_float(b);
}

// ---------------------------------------------------------------------------
// Prep: split fp32 embeddings into bf16 hi/lo arrays + row squared norms
// ---------------------------------------------------------------------------
__global__ void prep_kernel(const float* __restrict__ e,
                            unsigned short* __restrict__ ehi,
                            unsigned short* __restrict__ elo,
                            float* __restrict__ sq) {
  const int row = blockIdx.x;       // one block (256 threads) per row
  const int t   = threadIdx.x;
  const size_t idx = (size_t)row * DD + t;
  const float x = e[idx];
  const unsigned short h = f32_to_bf16_rne(x);
  const float hf = __uint_as_float(((unsigned)h) << 16);
  const unsigned short l = f32_to_bf16_rne(x - hf);
  ehi[idx] = h;
  elo[idx] = l;

  float p = x * x;
  #pragma unroll
  for (int off = 1; off < 32; off <<= 1) p += __shfl_xor(p, off, 32);
  __shared__ float part[8];
  if ((t & 31) == 0) part[t >> 5] = p;
  __syncthreads();
  if (t == 0) {
    float s = 0.0f;
    #pragma unroll
    for (int i = 0; i < 8; ++i) s += part[i];
    sq[row] = s;
  }
}

// ---------------------------------------------------------------------------
// Init the order-encoded reduction arrays
// ---------------------------------------------------------------------------
__global__ void init_kernel(unsigned* __restrict__ dapU,
                            unsigned* __restrict__ sminU,
                            unsigned* __restrict__ aminU) {
  const int i = blockIdx.x * 256 + threadIdx.x;
  if (i < BB) {
    dapU[i]  = f2ord(-BIGV);
    sminU[i] = f2ord(BIGV);
    aminU[i] = f2ord(BIGV);
  }
}

// ---------------------------------------------------------------------------
// One 16x16 Gram tile over K=256, bf16 hi/lo split, 3 independent WMMA chains
// ---------------------------------------------------------------------------
static __device__ __forceinline__ v8f tile_dot(const FragU (&Ah)[8],
                                               const FragU (&Al)[8],
                                               const unsigned short* __restrict__ cH,
                                               const unsigned short* __restrict__ cL,
                                               const int (&ko)[8]) {
  v8f a0 = {}, a1 = {}, a2 = {};
  #pragma unroll
  for (int c = 0; c < 8; ++c) {
    FragU Bh, Bl;
    #pragma unroll
    for (int r = 0; r < 8; ++r) {
      Bh.u[r] = *(const unsigned*)(cH + 32 * c + ko[r]);
      Bl.u[r] = *(const unsigned*)(cL + 32 * c + ko[r]);
    }
    a0 = wmma_bf16(Ah[c].v, Bh.v, a0);   // hi*hi chain
    a1 = wmma_bf16(Ah[c].v, Bl.v, a1);   // hi*lo chain
    a2 = wmma_bf16(Al[c].v, Bh.v, a2);   // lo*hi chain
  }
  v8f acc;
  #pragma unroll
  for (int r = 0; r < 8; ++r) acc[r] = a0[r] + (a1[r] + a2[r]);
  return acc;
}

// ---------------------------------------------------------------------------
// Symmetric-half pass kernel. Each block owns row strips {b, 511-b} (513 tiles
// total, balanced). Each 16x16 tile (j >= i) is applied to BOTH its row strip
// (register accumulation, merged once per strip) and its column strip
// (cross-half shfl + per-tile uint atomics). Max/min are idempotent, so any
// incidental duplication (diagonal tiles) is harmless, and the exact
// order-independence of max/min keeps results deterministic.
// PASS 1: hardest positive -> dapU.  PASS 2: negative mins -> aminU/sminU.
// ---------------------------------------------------------------------------
template <int PASS>
__global__ __launch_bounds__(256) void pass_kernel(
    const unsigned short* __restrict__ ehi,
    const unsigned short* __restrict__ elo,
    const float* __restrict__ sq,
    const int* __restrict__ labels,
    unsigned* __restrict__ dapU,
    unsigned* __restrict__ sminU,
    unsigned* __restrict__ aminU) {
  const int wave = threadIdx.x >> 5;   // 0..7
  const int lane = threadIdx.x & 31;
  const int half = lane >> 4;
  const int lsub = lane & 15;

  // K element offsets for the 16-bit 16x32 fragment layout (ISA 7.12.2)
  int ko[8];
  #pragma unroll
  for (int r = 0; r < 8; ++r)
    ko[r] = ((r < 4) ? (2 * r) : (16 + 2 * (r - 4))) + (half ? 8 : 0);

  #pragma unroll 1
  for (int sidx = 0; sidx < 2; ++sidx) {
    const int si = sidx ? (NTILE - 1 - (int)blockIdx.x) : (int)blockIdx.x;
    const int m0 = si * 16;

    // Resident A fragments for this strip: 16 anchor rows x K=256, hi+lo
    FragU Ah[8], Al[8];
    {
      const unsigned short* rH = ehi + (size_t)(m0 + lsub) * DD;
      const unsigned short* rL = elo + (size_t)(m0 + lsub) * DD;
      #pragma unroll
      for (int c = 0; c < 8; ++c) {
        #pragma unroll
        for (int r = 0; r < 8; ++r) {
          Ah[c].u[r] = *(const unsigned*)(rH + 32 * c + ko[r]);
          Al[c].u[r] = *(const unsigned*)(rL + 32 * c + ko[r]);
        }
      }
    }

    float sqm[8]; int labm[8]; float dapm[8];
    #pragma unroll
    for (int r = 0; r < 8; ++r) {
      const int m = m0 + r + half * 8;   // C/D layout: M = r + 8*(lane>=16)
      sqm[r]  = sq[m];
      labm[r] = labels[m];
      dapm[r] = (PASS == 2) ? ord2f(dapU[m]) : 0.0f;
    }

    // row-view accumulators: PASS1: racc0 = pos max.
    //                        PASS2: racc0 = all-neg min, racc1 = semi-hard min.
    float racc0[8], racc1[8];
    #pragma unroll
    for (int r = 0; r < 8; ++r) {
      racc0[r] = (PASS == 1) ? -BIGV : BIGV;
      racc1[r] = BIGV;
    }

    for (int jt = si + wave; jt < NTILE; jt += 8) {
      const int n0 = jt * 16;
      const int n  = n0 + lsub;
      const unsigned short* cH = ehi + (size_t)n * DD;
      const unsigned short* cL = elo + (size_t)n * DD;
      {
        const int np = ((jt + 8) & (NTILE - 1)) * 16 + lsub;
        __builtin_prefetch(ehi + (size_t)np * DD, 0, 3);   // global_prefetch_b8
      }
      const v8f acc = tile_dot(Ah, Al, cH, cL, ko);
      const float sqn  = sq[n];
      const int   labn = labels[n];
      const float dapn = (PASS == 2) ? ord2f(dapU[n]) : 0.0f;

      // column-view partials for this tile (per lane: fixed n, 8 m values)
      float colA = (PASS == 1) ? -BIGV : BIGV;
      float colS = BIGV;

      #pragma unroll
      for (int r = 0; r < 8; ++r) {
        const int m = m0 + r + half * 8;
        const float d = fmaxf(sqm[r] + sqn - 2.0f * acc[r], 0.0f);
        if (PASS == 1) {
          if ((labn == labm[r]) && (n != m)) {
            racc0[r] = fmaxf(racc0[r], d);       // row view: pos max
            colA     = fmaxf(colA, d);           // col view: pos max
          }
        } else {
          if (labn != labm[r]) {
            racc0[r] = fminf(racc0[r], d);       // row view: all-neg min
            colA     = fminf(colA, d);           // col view: all-neg min
            if (d > dapm[r] && d < dapm[r] + MARGINF) racc1[r] = fminf(racc1[r], d);
            if (d > dapn    && d < dapn    + MARGINF) colS     = fminf(colS, d);
          }
        }
      }

      // fold the two halves (lanes L and L^16 share n = lsub), low lanes commit
      if (PASS == 1) {
        colA = fmaxf(colA, __shfl_xor(colA, 16, 32));
        if (half == 0 && colA > -0.5f * BIGV) atomicMax(&dapU[n], f2ord(colA));
      } else {
        colA = fminf(colA, __shfl_xor(colA, 16, 32));
        colS = fminf(colS, __shfl_xor(colS, 16, 32));
        if (half == 0) {
          if (colA < 0.5f * BIGV) atomicMin(&aminU[n], f2ord(colA));
          if (colS < 0.5f * BIGV) atomicMin(&sminU[n], f2ord(colS));
        }
      }
    }

    // strip-end row merge: butterfly over the 16 lanes sharing row M
    #pragma unroll
    for (int r = 0; r < 8; ++r) {
      if (PASS == 1) {
        racc0[r] = fmaxf(racc0[r], __shfl_xor(racc0[r], 1, 32));
        racc0[r] = fmaxf(racc0[r], __shfl_xor(racc0[r], 2, 32));
        racc0[r] = fmaxf(racc0[r], __shfl_xor(racc0[r], 4, 32));
        racc0[r] = fmaxf(racc0[r], __shfl_xor(racc0[r], 8, 32));
      } else {
        racc0[r] = fminf(racc0[r], __shfl_xor(racc0[r], 1, 32));
        racc0[r] = fminf(racc0[r], __shfl_xor(racc0[r], 2, 32));
        racc0[r] = fminf(racc0[r], __shfl_xor(racc0[r], 4, 32));
        racc0[r] = fminf(racc0[r], __shfl_xor(racc0[r], 8, 32));
        racc1[r] = fminf(racc1[r], __shfl_xor(racc1[r], 1, 32));
        racc1[r] = fminf(racc1[r], __shfl_xor(racc1[r], 2, 32));
        racc1[r] = fminf(racc1[r], __shfl_xor(racc1[r], 4, 32));
        racc1[r] = fminf(racc1[r], __shfl_xor(racc1[r], 8, 32));
      }
    }
    if (lsub == 0) {
      #pragma unroll
      for (int r = 0; r < 8; ++r) {
        const int m = m0 + r + half * 8;
        if (PASS == 1) {
          if (racc0[r] > -0.5f * BIGV) atomicMax(&dapU[m], f2ord(racc0[r]));
        } else {
          if (racc0[r] < 0.5f * BIGV) atomicMin(&aminU[m], f2ord(racc0[r]));
          if (racc1[r] < 0.5f * BIGV) atomicMin(&sminU[m], f2ord(racc1[r]));
        }
      }
    }
  }
}

// ---------------------------------------------------------------------------
// Deterministic finalize: per-row loss from the three reduction arrays
// ---------------------------------------------------------------------------
__global__ void finalize_kernel(const unsigned* __restrict__ dapU,
                                const unsigned* __restrict__ sminU,
                                const unsigned* __restrict__ aminU,
                                float* __restrict__ out) {
  float sum = 0.0f, cnt = 0.0f;
  for (int i = 0; i < BB; ++i) {
    const float dap = ord2f(dapU[i]);
    const float s   = ord2f(sminU[i]);
    const float a   = ord2f(aminU[i]);
    const float dan = (s < 0.5f * BIGV) ? s : a;   // semi-hard, else all-neg
    if (dap > -0.5f * BIGV) {                      // has_pos
      sum += fmaxf(dap - dan + MARGINF, 0.0f);
      cnt += 1.0f;
    }
  }
  out[0] = sum / fmaxf(cnt, 1.0f);
}

// ---------------------------------------------------------------------------
extern "C" void kernel_launch(void* const* d_in, const int* in_sizes, int n_in,
                              void* d_out, int out_size, void* d_ws, size_t ws_size,
                              hipStream_t stream) {
  (void)in_sizes; (void)n_in; (void)out_size; (void)ws_size;
  const float* emb    = (const float*)d_in[0];
  const int*   labels = (const int*)d_in[1];
  float*       out    = (float*)d_out;

  char* w = (char*)d_ws;
  unsigned short* ehi = (unsigned short*)w;                            // 4 MB
  unsigned short* elo = (unsigned short*)(w + (size_t)BB * DD * 2);    // 4 MB
  char* base8 = w + (size_t)BB * DD * 4;                               // +8 MB
  float*    sq    = (float*)base8;                                     // 32 KB
  unsigned* dapU  = (unsigned*)(base8 + (size_t)BB * 4);               // 32 KB
  unsigned* sminU = (unsigned*)(base8 + (size_t)BB * 8);               // 32 KB
  unsigned* aminU = (unsigned*)(base8 + (size_t)BB * 12);              // 32 KB

  prep_kernel<<<BB, DD, 0, stream>>>(emb, ehi, elo, sq);
  init_kernel<<<BB / 256, 256, 0, stream>>>(dapU, sminU, aminU);
  pass_kernel<1><<<NSTRIP, 256, 0, stream>>>(ehi, elo, sq, labels,
                                             dapU, sminU, aminU);
  pass_kernel<2><<<NSTRIP, 256, 0, stream>>>(ehi, elo, sq, labels,
                                             dapU, sminU, aminU);
  finalize_kernel<<<1, 1, 0, stream>>>(dapU, sminU, aminU, out);
}
